// PoolingAttentionNetLayer_3616362463819
// MI455X (gfx1250) — matrix operand order repaired and checked by reading, not compiled, not run on hardware
//
#include <hip/hip_runtime.h>
#include <math.h>

// ---------------- problem constants (from reference) ----------------
#define B_    8
#define N_    4096
#define CP    64
#define S_    1024
#define KNB   32
#define M_    (B_ * S_ * KNB)     // 262144 grouped rows
#define NG    (B_ * S_)           // 8192 groups
#define RAD2  (0.2f * 0.2f)
#define DK    128
#define DOUT  128
#define BN_EPS 1e-3f

typedef _Float16 v16h __attribute__((ext_vector_type(16)));
typedef _Float16 v8h  __attribute__((ext_vector_type(8)));
typedef float    v8f  __attribute__((ext_vector_type(8)));

// ---------------------------------------------------------------
// 1) Farthest point sampling: one block per batch, 1024 serial
//    argmax reductions. dist kept in registers (16 pts / thread).
// ---------------------------------------------------------------
__global__ void __launch_bounds__(256)
fps_kernel(const float* __restrict__ xyz, float* __restrict__ new_xyz) {
    const int b = blockIdx.x;
    const int t = threadIdx.x;
    const float* xb = xyz + (size_t)b * N_ * 3;

    __shared__ float cen[3];
    __shared__ float rv[256];
    __shared__ int   ri[256];
    __shared__ int   s_far;

    float dist[N_ / 256];
#pragma unroll
    for (int j = 0; j < N_ / 256; ++j) dist[j] = 1e10f;
    if (t == 0) s_far = 0;
    __syncthreads();

    for (int s = 0; s < S_; ++s) {
        const int far = s_far;
        if (t < 3) cen[t] = xb[far * 3 + t];
        __syncthreads();
        const float cx = cen[0], cy = cen[1], cz = cen[2];

        float bm = -1.0f; int bi = 0;
#pragma unroll
        for (int j = 0; j < N_ / 256; ++j) {
            const int i = j * 256 + t;
            const float dx = xb[i * 3 + 0] - cx;
            const float dy = xb[i * 3 + 1] - cy;
            const float dz = xb[i * 3 + 2] - cz;
            const float d  = dx * dx + dy * dy + dz * dz;
            const float nd = fminf(dist[j], d);
            dist[j] = nd;
            if (nd > bm) { bm = nd; bi = i; }
        }
        rv[t] = bm; ri[t] = bi;
        __syncthreads();
        for (int off = 128; off > 0; off >>= 1) {
            if (t < off) {
                const float ov = rv[t + off]; const int oi = ri[t + off];
                if (ov > rv[t] || (ov == rv[t] && oi < ri[t])) { rv[t] = ov; ri[t] = oi; }
            }
            __syncthreads();
        }
        if (t == 0) {
            new_xyz[((size_t)b * S_ + s) * 3 + 0] = cen[0];
            new_xyz[((size_t)b * S_ + s) * 3 + 1] = cen[1];
            new_xyz[((size_t)b * S_ + s) * 3 + 2] = cen[2];
            s_far = ri[0];
        }
        __syncthreads();
    }
}

// ---------------------------------------------------------------
// 2) Ball query: one wave per group; first KNB in-radius indices
//    in index order via ballot/popcount rank, padded with slot 0.
// ---------------------------------------------------------------
__global__ void __launch_bounds__(256)
ballq_kernel(const float* __restrict__ xyz, const float* __restrict__ new_xyz,
             int* __restrict__ idx_i, float* __restrict__ idx_f) {
    const int wid  = threadIdx.x >> 5;
    const int lane = threadIdx.x & 31;
    const int g    = blockIdx.x * 8 + wid;        // 0..NG-1
    const int b    = g >> 10;
    const float* xb = xyz + (size_t)b * N_ * 3;

    const float cx = new_xyz[(size_t)g * 3 + 0];
    const float cy = new_xyz[(size_t)g * 3 + 1];
    const float cz = new_xyz[(size_t)g * 3 + 2];

    __shared__ int slots[8][KNB];
    int cnt = 0;
    for (int base = 0; base < N_ && cnt < KNB; base += 32) {
        const int i = base + lane;
        const float dx = xb[i * 3 + 0] - cx;
        const float dy = xb[i * 3 + 1] - cy;
        const float dz = xb[i * 3 + 2] - cz;
        const bool inb = (dx * dx + dy * dy + dz * dz) < RAD2;
        const unsigned mask = (unsigned)__ballot(inb);
        if (inb) {
            const int rank = __popc(mask & ((1u << lane) - 1u));
            const int slot = cnt + rank;
            if (slot < KNB) slots[wid][slot] = i;
        }
        cnt += __popc(mask);
        if (cnt > KNB) cnt = KNB;
    }
    __syncthreads();
    const int v0 = slots[wid][0];                 // center always in ball -> cnt>=1
    const int v  = (lane < cnt) ? slots[wid][lane] : v0;
    idx_i[(size_t)g * KNB + lane] = v;
    idx_f[(size_t)g * KNB + lane] = (float)v;
}

// ---------------------------------------------------------------
// 3) Gather + concat(local xyz, point feats), zero-pad 67 -> 96,
//    emit f16 rows for the WMMA GEMM chain.
// ---------------------------------------------------------------
__global__ void gather_kernel(const float* __restrict__ xyz, const float* __restrict__ points,
                              const float* __restrict__ new_xyz, const int* __restrict__ idx_i,
                              _Float16* __restrict__ X) {
    const int KP = 96;
    const size_t total = (size_t)M_ * KP;
    for (size_t e = (size_t)blockIdx.x * blockDim.x + threadIdx.x; e < total;
         e += (size_t)gridDim.x * blockDim.x) {
        const int    c   = (int)(e % KP);
        const size_t row = e / KP;
        const int    k   = (int)(row % KNB);
        const size_t g   = row / KNB;
        const int    b   = (int)(g >> 10);
        const int    pi  = idx_i[g * KNB + k];
        float val;
        if (c < 3)            val = xyz[((size_t)b * N_ + pi) * 3 + c] - new_xyz[g * 3 + c];
        else if (c < 3 + CP)  val = points[((size_t)b * N_ + pi) * CP + (c - 3)];
        else                  val = 0.0f;
        X[e] = (_Float16)val;
    }
}

// Weight prep: W[Kin,Nd] f32 -> Wt[Nd,Kp] f16 (transposed, zero-padded K)
__global__ void wprep_kernel(const float* __restrict__ W, _Float16* __restrict__ Wt,
                             int Kin, int Nd, int Kp) {
    const int total = Nd * Kp;
    for (int e = blockIdx.x * blockDim.x + threadIdx.x; e < total; e += gridDim.x * blockDim.x) {
        const int k = e % Kp;
        const int n = e / Kp;
        Wt[e] = (_Float16)((k < Kin) ? W[(size_t)k * Nd + n] : 0.0f);
    }
}

__global__ void zero_kernel(float* __restrict__ p, int n) {
    const int i = blockIdx.x * blockDim.x + threadIdx.x;
    if (i < n) p[i] = 0.0f;
}

// ---------------------------------------------------------------
// 4) WMMA GEMM: Y[M, TN*16] = X[M,Kd] @ Wt^T (+bias).
//    One wave owns a 16-row strip x ALL column tiles: the A
//    fragment is loaded once per K-step and reused across TN
//    v_wmma_f32_16x16x32_f16 issues (TN x A-traffic reduction,
//    TN WMMAs per (2*TN+2) b128 loads). Optional fused per-channel
//    sum/sumsq (BN stats) via shfl_xor(16) + atomicAdd epilogue.
// ---------------------------------------------------------------
template<int TN>
__global__ void __launch_bounds__(256)
gemm_wmma_kernel(const _Float16* __restrict__ X, const _Float16* __restrict__ Wt,
                 const float* __restrict__ bias, _Float16* __restrict__ Yh,
                 float* __restrict__ stats, int Mrows, int Kd) {
    const int Nd   = TN * 16;
    const int wave = (blockIdx.x * blockDim.x + threadIdx.x) >> 5;   // 16-row strip id
    const int lane = threadIdx.x & 31;
    if (wave * 16 >= Mrows) return;

    const int h   = lane >> 4;       // half-wave select
    const int l15 = lane & 15;

    v8f acc[TN];
#pragma unroll
    for (int tn = 0; tn < TN; ++tn) {
        const float bv = bias ? bias[tn * 16 + l15] : 0.0f;
#pragma unroll
        for (int j = 0; j < 8; ++j) acc[tn][j] = bv;
    }

    // A lane layout (16b A 16x32): elems 0-7 -> K=8h+0..7, elems 8-15 -> K=16+8h+0..7
    const _Float16* arow  = X  + (size_t)(wave * 16 + l15) * Kd + 8 * h;
    // B lane layout: elems 0-7 -> K=8h+0..7, elems 8-15 -> K=16+8h+0..7 for column (tn*16 + l15); Wt is [Nd,Kd]
    const _Float16* bbase = Wt + (size_t)l15 * Kd + 8 * h;

    for (int k0 = 0; k0 < Kd; k0 += 32) {
        union { v16h v; v8h p[2]; } a;
        a.p[0] = *(const v8h*)(arow + k0);
        a.p[1] = *(const v8h*)(arow + k0 + 16);
#pragma unroll
        for (int tn = 0; tn < TN; ++tn) {
            union { v16h v; v8h p[2]; } bm;
            const _Float16* bp = bbase + (size_t)tn * 16 * Kd + k0;
            bm.p[0] = *(const v8h*)(bp);
            bm.p[1] = *(const v8h*)(bp + 16);
            acc[tn] = __builtin_amdgcn_wmma_f32_16x16x32_f16(false, a.v, false, bm.v,
                                                             (short)0, acc[tn], false, false);
        }
    }

    // store: C/D layout -> m = 16*wave + 8h + j, col tn*16 + l15
    const size_t mbase = (size_t)wave * 16 + 8 * h;
#pragma unroll
    for (int tn = 0; tn < TN; ++tn) {
        const int n = tn * 16 + l15;
#pragma unroll
        for (int j = 0; j < 8; ++j)
            Yh[(mbase + j) * Nd + n] = (_Float16)acc[tn][j];
        if (stats) {
            float s = 0.0f, q = 0.0f;
#pragma unroll
            for (int j = 0; j < 8; ++j) { s += acc[tn][j]; q += acc[tn][j] * acc[tn][j]; }
            s += __shfl_xor(s, 16, 32);
            q += __shfl_xor(q, 16, 32);
            if (h == 0) {
                atomicAdd(&stats[n], s);
                atomicAdd(&stats[Nd + n], q);
            }
        }
    }
}

// BN finalize: per-channel scale/shift from accumulated sum / sumsq
__global__ void bnfin_kernel(const float* __restrict__ stats, const float* __restrict__ g,
                             const float* __restrict__ be, float* __restrict__ scale,
                             float* __restrict__ shift, int Nd, float invM) {
    const int c = threadIdx.x;
    if (c < Nd) {
        const float mean = stats[c] * invM;
        const float var  = stats[Nd + c] * invM - mean * mean;
        const float sc   = g[c] * rsqrtf(var + BN_EPS);
        scale[c] = sc;
        shift[c] = be[c] - mean * sc;
    }
}

// fused BN + ReLU + f16 requantize into next layer's input
__global__ void bnrelu_kernel(const _Float16* __restrict__ Y, const float* __restrict__ scale,
                              const float* __restrict__ shift, _Float16* __restrict__ X,
                              int Nd, size_t total) {
    for (size_t e = (size_t)blockIdx.x * blockDim.x + threadIdx.x; e < total;
         e += (size_t)gridDim.x * blockDim.x) {
        const int c = (int)(e % Nd);
        const float o = (float)Y[e] * scale[c] + shift[c];
        X[e] = (_Float16)fmaxf(o, 0.0f);
    }
}

// q = new_xyz @ Wq  (K=3, tiny -> plain VALU)
__global__ void qproj_kernel(const float* __restrict__ new_xyz, const float* __restrict__ Wq,
                             float* __restrict__ q) {
    const size_t total = (size_t)NG * DK;
    for (size_t e = (size_t)blockIdx.x * blockDim.x + threadIdx.x; e < total;
         e += (size_t)gridDim.x * blockDim.x) {
        const int    d = (int)(e & (DK - 1));
        const size_t g = e >> 7;
        float acc = 0.0f;
#pragma unroll
        for (int c = 0; c < 3; ++c) acc += new_xyz[g * 3 + c] * Wq[c * DK + d];
        q[e] = acc;
    }
}

// ---------------------------------------------------------------
// 5) Attention pooling: one 128-thread block per group.
//    scores = softmax(q . k / sqrt(dk)); pooled = scores @ v
// ---------------------------------------------------------------
__global__ void __launch_bounds__(128)
attn_kernel(const float* __restrict__ q, const _Float16* __restrict__ Kf,
            const _Float16* __restrict__ Vf, float* __restrict__ pooled) {
    const int g    = blockIdx.x;          // group
    const int t    = threadIdx.x;         // 0..127 = feature dim
    const int lane = t & 31, wid = t >> 5;

    __shared__ float sc[KNB];
    __shared__ float red[4];

    const float qv = q[(size_t)g * DK + t];
    const _Float16* kb = Kf + (size_t)g * KNB * DK;

    for (int k = 0; k < KNB; ++k) {
        float p = qv * (float)kb[k * DK + t];
#pragma unroll
        for (int o = 16; o > 0; o >>= 1) p += __shfl_xor(p, o, 32);
        if (lane == 0) red[wid] = p;
        __syncthreads();
        if (t == 0) sc[k] = (red[0] + red[1] + red[2] + red[3]) * 0.08838834764831845f; // 1/sqrt(128)
        __syncthreads();
    }
    if (wid == 0) {
        const float v = sc[lane];
        float mx = v;
#pragma unroll
        for (int o = 16; o > 0; o >>= 1) mx = fmaxf(mx, __shfl_xor(mx, o, 32));
        const float e = expf(v - mx);
        float ssum = e;
#pragma unroll
        for (int o = 16; o > 0; o >>= 1) ssum += __shfl_xor(ssum, o, 32);
        sc[lane] = e / ssum;
    }
    __syncthreads();
    const _Float16* vb = Vf + (size_t)g * KNB * DK;
    float acc = 0.0f;
    for (int k = 0; k < KNB; ++k) acc += sc[k] * (float)vb[k * DK + t];
    pooled[(size_t)g * DOUT + t] = acc;
}

// ---------------------------------------------------------------
extern "C" void kernel_launch(void* const* d_in, const int* in_sizes, int n_in,
                              void* d_out, int out_size, void* d_ws, size_t ws_size,
                              hipStream_t stream) {
    (void)in_sizes; (void)n_in; (void)out_size; (void)ws_size;
    const float* xyz    = (const float*)d_in[0];
    const float* points = (const float*)d_in[1];
    const float* W0 = (const float*)d_in[2],  *b0 = (const float*)d_in[3];
    const float* g0 = (const float*)d_in[4],  *be0 = (const float*)d_in[5];
    const float* W1 = (const float*)d_in[6],  *b1 = (const float*)d_in[7];
    const float* g1 = (const float*)d_in[8],  *be1 = (const float*)d_in[9];
    const float* W2 = (const float*)d_in[10], *b2 = (const float*)d_in[11];
    const float* g2 = (const float*)d_in[12], *be2 = (const float*)d_in[13];
    const float* Wq = (const float*)d_in[14];
    const float* Wk = (const float*)d_in[15];
    const float* Wv = (const float*)d_in[16];

    float* out         = (float*)d_out;
    float* out_newxyz  = out;                               // NG*3
    float* out_pooled  = out + (size_t)NG * 3;              // NG*128
    float* out_idx     = out + (size_t)NG * 3 + (size_t)NG * DOUT; // NG*32 (int as float)

    char* ws = (char*)d_ws;
    size_t off = 0;
    auto alloc = [&](size_t bytes) -> void* {
        void* p = ws + off;
        off += (bytes + 255) & ~(size_t)255;
        return p;
    };
    _Float16* X16  = (_Float16*)alloc((size_t)M_ * 128 * 2);   // activations (96/64/128 cols)
    _Float16* Y16  = (_Float16*)alloc((size_t)M_ * 128 * 2);   // GEMM out; reused as K-matrix
    _Float16* V16  = (_Float16*)alloc((size_t)M_ * 128 * 2);   // V-matrix
    float*    qbuf = (float*)alloc((size_t)NG * DK * 4);
    int*      idxi = (int*)alloc((size_t)NG * KNB * 4);
    float*    stats = (float*)alloc(256 * 4);
    float*    scale = (float*)alloc(128 * 4);
    float*    shift = (float*)alloc(128 * 4);
    _Float16* W0t = (_Float16*)alloc(64 * 96 * 2);
    _Float16* W1t = (_Float16*)alloc(64 * 64 * 2);
    _Float16* W2t = (_Float16*)alloc(128 * 64 * 2);
    _Float16* Wkt = (_Float16*)alloc(128 * 128 * 2);
    _Float16* Wvt = (_Float16*)alloc(128 * 128 * 2);

    const float invM = 1.0f / (float)M_;
    const int gemm_blocks = (M_ / 16) / 8;   // 8 waves (16-row strips) per 256-thread block

    // weight prep
    wprep_kernel<<<32, 256, 0, stream>>>(W0, W0t, 67, 64, 96);
    wprep_kernel<<<32, 256, 0, stream>>>(W1, W1t, 64, 64, 64);
    wprep_kernel<<<64, 256, 0, stream>>>(W2, W2t, 64, 128, 64);
    wprep_kernel<<<64, 256, 0, stream>>>(Wk, Wkt, 128, 128, 128);
    wprep_kernel<<<64, 256, 0, stream>>>(Wv, Wvt, 128, 128, 128);

    // sampling / grouping
    fps_kernel<<<B_, 256, 0, stream>>>(xyz, out_newxyz);
    ballq_kernel<<<NG / 8, 256, 0, stream>>>(xyz, out_newxyz, idxi, out_idx);
    gather_kernel<<<8192, 256, 0, stream>>>(xyz, points, out_newxyz, idxi, X16);

    // MLP layer 0: [M,96] @ [96,64]
    zero_kernel<<<1, 256, 0, stream>>>(stats, 256);
    gemm_wmma_kernel<4><<<gemm_blocks, 256, 0, stream>>>(X16, W0t, b0, Y16, stats, M_, 96);
    bnfin_kernel<<<1, 128, 0, stream>>>(stats, g0, be0, scale, shift, 64, invM);
    bnrelu_kernel<<<8192, 256, 0, stream>>>(Y16, scale, shift, X16, 64, (size_t)M_ * 64);

    // MLP layer 1: [M,64] @ [64,64]
    zero_kernel<<<1, 256, 0, stream>>>(stats, 256);
    gemm_wmma_kernel<4><<<gemm_blocks, 256, 0, stream>>>(X16, W1t, b1, Y16, stats, M_, 64);
    bnfin_kernel<<<1, 128, 0, stream>>>(stats, g1, be1, scale, shift, 64, invM);
    bnrelu_kernel<<<8192, 256, 0, stream>>>(Y16, scale, shift, X16, 64, (size_t)M_ * 64);

    // MLP layer 2: [M,64] @ [64,128]
    zero_kernel<<<1, 256, 0, stream>>>(stats, 256);
    gemm_wmma_kernel<8><<<gemm_blocks, 256, 0, stream>>>(X16, W2t, b2, Y16, stats, M_, 64);
    bnfin_kernel<<<1, 128, 0, stream>>>(stats, g2, be2, scale, shift, 128, invM);
    bnrelu_kernel<<<8192, 256, 0, stream>>>(Y16, scale, shift, X16, 128, (size_t)M_ * 128);

    // attention projections: K into Y16, V into V16 (no bias, no BN)
    gemm_wmma_kernel<8><<<gemm_blocks, 256, 0, stream>>>(X16, Wkt, nullptr, Y16, nullptr, M_, 128);
    gemm_wmma_kernel<8><<<gemm_blocks, 256, 0, stream>>>(X16, Wvt, nullptr, V16, nullptr, M_, 128);

    // q projection + attention pooling
    qproj_kernel<<<4096, 256, 0, stream>>>(out_newxyz, Wq, qbuf);
    attn_kernel<<<NG, 128, 0, stream>>>(qbuf, Y16, V16, out_pooled);
}
